// SparseMoE_53841710022746
// MI455X (gfx1250) — matrix operand order repaired and checked by reading, not compiled, not run on hardware
//
#include <hip/hip_runtime.h>
#include <hip/hip_bf16.h>
#include <math.h>

// ---------------------------------------------------------------------------
// Types for CDNA5 WMMA / TDM
// ---------------------------------------------------------------------------
typedef __attribute__((ext_vector_type(16))) __bf16       v16bf;
typedef __attribute__((ext_vector_type(8)))  float        v8f;
typedef __attribute__((ext_vector_type(4)))  unsigned int u32x4;
typedef __attribute__((ext_vector_type(8)))  int          i32x8;
typedef __attribute__((ext_vector_type(4)))  int          i32x4;

#define NEXP 8
#define TOPK 2

__device__ __forceinline__ float gelu_exact(float x) {
    return 0.5f * x * (1.0f + erff(x * 0.70710678118654752f));
}

// Load a 16-element bf16 WMMA fragment as two 16B LDS reads.
__device__ __forceinline__ v16bf ld_frag(const __bf16* p0, const __bf16* p1) {
    union { v16bf v; uint4 u[2]; } f;
    f.u[0] = *(const uint4*)p0;
    f.u[1] = *(const uint4*)p1;
    return f.v;
}

// ---------------------------------------------------------------------------
// TDM: async 2D bf16 tile load Global -> LDS with row padding.
// D# per CDNA5 ISA ch.8: group0 = {count/type/lds_addr/global_addr},
// group1 = {data_size, pad, tensor/tile dims, dim0 stride}. 2D tensor ->
// groups 2/3 zero. tensor_dims == tile dims so no OOB path is exercised.
//   padIntervalCode 3 -> pad after every 16 DWORDs (64B = one 32-elem row)
//   padAmountCode   3 -> insert 4 DWORDs (16B) => 80B row stride (LDA=40)
// ---------------------------------------------------------------------------
__device__ __forceinline__ void tdm_load_2d_bf16(
    unsigned ldsAddr, const __bf16* gsrc, int tileX, int tileY,
    long long strideElems, int padIntervalCode, int padAmountCode)
{
    unsigned long long ga = (unsigned long long)(uintptr_t)gsrc;
    u32x4 g0;
    g0.x = 1u;                                           // count=1 (valid), user mode
    g0.y = ldsAddr;                                      // LDS byte address
    g0.z = (unsigned)ga;                                 // global_addr[31:0]
    g0.w = (unsigned)((ga >> 32) & 0x01FFFFFFu) | (2u << 30);  // addr[56:32] | type=2
    unsigned td0 = (unsigned)tileX, td1 = (unsigned)tileY;
    unsigned long long s0 = (unsigned long long)strideElems;
    i32x8 g1;
    g1[0] = (int)((1u << 16)                              // data_size=1 (2 bytes)
                | (1u << 20)                              // pad_enable
                | ((unsigned)padIntervalCode << 22)
                | ((unsigned)padAmountCode << 25));
    g1[1] = (int)((td0 & 0xFFFFu) << 16);                 // tensor_dim0[15:0]
    g1[2] = (int)((td0 >> 16) | ((td1 & 0xFFFFu) << 16)); // tensor_dim0 hi | tensor_dim1 lo
    g1[3] = (int)((td1 >> 16) | ((unsigned)tileX << 16)); // tensor_dim1 hi | tile_dim0
    g1[4] = (int)(unsigned)tileY;                         // tile_dim1 (tile_dim2 = 0)
    g1[5] = (int)(unsigned)(s0 & 0xFFFFFFFFull);          // tensor_dim0_stride lo
    g1[6] = (int)(unsigned)((s0 >> 32) & 0xFFFFull);      // stride hi | dim1_stride lo (0)
    g1[7] = 0;
    i32x4 gz = (i32x4){0, 0, 0, 0};
#if defined(__clang_major__) && (__clang_major__ >= 23)
    i32x8 gz8 = (i32x8){0, 0, 0, 0, 0, 0, 0, 0};
    __builtin_amdgcn_tensor_load_to_lds(g0, g1, gz, gz, gz8, 0);
#else
    __builtin_amdgcn_tensor_load_to_lds(g0, g1, gz, gz, 0);
#endif
}

// ---------------------------------------------------------------------------
// Kernel 1: router — logits = x @ Wg^T, softmax over 8, top-2.
// One wave (32 lanes) per token; lanes split the d dimension.
// ---------------------------------------------------------------------------
__global__ __launch_bounds__(256) void router_kernel(
    const float* __restrict__ flat, const float* __restrict__ Wg,
    int* __restrict__ slot_expert, float* __restrict__ slot_score,
    int d, int N)
{
    __shared__ float WgL[NEXP * 1024];
    const int tid = threadIdx.x;
    for (int i = tid; i < NEXP * d; i += blockDim.x) WgL[i] = Wg[i];
    __syncthreads();

    const int wave = tid >> 5;
    const int lane = tid & 31;
    const int n = blockIdx.x * 8 + wave;
    if (n >= N) return;

    float acc[NEXP];
#pragma unroll
    for (int e = 0; e < NEXP; ++e) acc[e] = 0.0f;

    const float* xr = flat + (size_t)n * d;
    for (int i = lane; i < d; i += 32) {
        float x = xr[i];
#pragma unroll
        for (int e = 0; e < NEXP; ++e) acc[e] += x * WgL[e * d + i];
    }
#pragma unroll
    for (int e = 0; e < NEXP; ++e) {
#pragma unroll
        for (int off = 16; off > 0; off >>= 1)
            acc[e] += __shfl_xor(acc[e], off, 32);
    }

    if (lane == 0) {
        // top-1 / top-2 on logits (softmax is monotonic; ties -> lower index)
        float m1 = acc[0]; int i1 = 0;
#pragma unroll
        for (int e = 1; e < NEXP; ++e) if (acc[e] > m1) { m1 = acc[e]; i1 = e; }
        float m2 = -INFINITY; int i2 = 0;
#pragma unroll
        for (int e = 0; e < NEXP; ++e)
            if (e != i1 && acc[e] > m2) { m2 = acc[e]; i2 = e; }
        float s = 0.0f;
#pragma unroll
        for (int e = 0; e < NEXP; ++e) s += expf(acc[e] - m1);
        float g1 = 1.0f / s;             // exp(0)/s
        float g2 = expf(m2 - m1) / s;
        slot_expert[2 * n + 0] = i1; slot_score[2 * n + 0] = g1;
        slot_expert[2 * n + 1] = i2; slot_score[2 * n + 1] = g2;
    }
}

// ---------------------------------------------------------------------------
// Kernel 2: capacity-limited dispatch. One block per expert.
// Exact top-`cap` by 4-pass radix select on positive-float bit patterns,
// then compaction with LDS atomic counters.
// ---------------------------------------------------------------------------
__global__ __launch_bounds__(1024) void dispatch_kernel(
    const int* __restrict__ slot_expert, const float* __restrict__ slot_score,
    int* __restrict__ tokList, float* __restrict__ gateList,
    int NK, int cap)
{
    const int e = blockIdx.x;
    const int tid = threadIdx.x;
    __shared__ unsigned int hist[256];
    __shared__ unsigned int sh_prefix, sh_k, sh_keepall, sh_pos, sh_eq;

    if (tid == 0) { sh_prefix = 0u; sh_k = (unsigned)cap; sh_keepall = 0u; sh_pos = 0u; sh_eq = 0u; }
    __syncthreads();

    for (int p = 3; p >= 0; --p) {
        if (tid < 256) hist[tid] = 0u;
        __syncthreads();
        unsigned prefix = sh_prefix;
        unsigned keep = sh_keepall;
        unsigned hmask = (p == 3) ? 0u : (0xFFFFFFFFu << ((p + 1) * 8));
        if (!keep) {
            for (int s = tid; s < NK; s += 1024) {
                if (slot_expert[s] == e) {
                    unsigned bits = __float_as_uint(slot_score[s]);
                    if ((bits & hmask) == prefix)
                        atomicAdd(&hist[(bits >> (p * 8)) & 255u], 1u);
                }
            }
        }
        __syncthreads();
        if (tid == 0 && !sh_keepall) {
            unsigned k = sh_k; int b;
            for (b = 255; b >= 0; --b) {
                if (hist[b] >= k) break;
                k -= hist[b];
            }
            if (b < 0) sh_keepall = 1u;               // fewer than cap candidates
            else { sh_prefix |= ((unsigned)b) << (p * 8); sh_k = k; }
        }
        __syncthreads();
    }

    // default-fill (unfilled slots: token 0, gate 0 -> zero contribution)
    for (int i = tid; i < cap; i += 1024) {
        tokList[(size_t)e * cap + i] = 0;
        gateList[(size_t)e * cap + i] = 0.0f;
    }
    __syncthreads();

    const unsigned T = sh_prefix;
    const unsigned quota = sh_k;
    const unsigned keep = sh_keepall;
    for (int s = tid; s < NK; s += 1024) {
        if (slot_expert[s] == e) {
            float sc = slot_score[s];
            unsigned bits = __float_as_uint(sc);
            bool ok = keep || (bits > T);
            if (!ok && bits == T) ok = (atomicAdd(&sh_eq, 1u) < quota);
            if (ok) {
                unsigned pidx = atomicAdd(&sh_pos, 1u);
                if (pidx < (unsigned)cap) {
                    tokList[(size_t)e * cap + pidx] = s >> 1;   // token = slot / K
                    gateList[(size_t)e * cap + pidx] = sc;
                }
            }
        }
    }
}

// ---------------------------------------------------------------------------
// Kernels 3/4: expert GEMMs with bf16 WMMA.
// Block tile 128x128, BK=32, 8 waves; each wave computes a 32x64 tile as
// 2x4 v_wmma_f32_16x16x32_bf16 accumulators.
//   GEMM1: A = gather(flat)[cap x d] (fp32->bf16 on load), B = W1[d x h],
//          epilogue bias + GELU -> bf16 hdn.
//   GEMM2: A = hdn[cap x h] (bf16) staged by the Tensor Data Mover (async,
//          TENSORcnt, row-padding to LDA), B = W2[h x d],
//          epilogue bias, * gate, atomicAdd into out[token].
// ---------------------------------------------------------------------------
#define LDA 40   // padded row stride (bf16 elems), 80B rows: 16B aligned, bank-spread
#define LDB 40

template <bool GEMM1>
__global__ __launch_bounds__(256) void moe_gemm_kernel(
    const float* __restrict__ flat,        // GEMM1 A source (token rows)
    const __bf16* __restrict__ hdnA,       // GEMM2 A source
    const float* __restrict__ W,           // [E, K, N] row-major
    const float* __restrict__ bias,        // [E, N]
    const int* __restrict__ tokList,       // [E, cap]
    const float* __restrict__ gateList,    // [E, cap]
    __bf16* __restrict__ hdnOut,           // GEMM1 destination
    float* __restrict__ out,               // GEMM2 destination [Ntok, N]
    int cap, int Ncols, int Kdim)
{
    const int e  = blockIdx.z;
    const int n0 = blockIdx.x * 128;
    const int m0 = blockIdx.y * 128;
    const int tid = threadIdx.x;

    __shared__ __bf16 As[128 * LDA];
    __shared__ __bf16 Bs[128 * LDB];
    __shared__ int   toks[128];
    __shared__ float gts[128];

    if (tid < 128) {
        toks[tid] = tokList[(size_t)e * cap + m0 + tid];
        gts[tid]  = gateList[(size_t)e * cap + m0 + tid];
    }
    __syncthreads();

    const float* We = W + (size_t)e * Kdim * Ncols;

    v8f acc[2][4];
#pragma unroll
    for (int i = 0; i < 2; ++i)
#pragma unroll
        for (int j = 0; j < 4; ++j)
            acc[i][j] = (v8f){0.f, 0.f, 0.f, 0.f, 0.f, 0.f, 0.f, 0.f};

    const int wave = tid >> 5;
    const int lane = tid & 31;
    const int wm = wave & 3;        // 4 waves along M (32 rows each)
    const int wn = wave >> 2;       // 2 waves along N (64 cols each)
    const int half = lane >> 4;     // K-half selector per ISA layout
    const int lc = lane & 15;

    for (int k0 = 0; k0 < Kdim; k0 += 32) {
        // ---- stage A tile [128 x 32] into LDS ----
        if (GEMM1) {
#pragma unroll
            for (int it = 0; it < 4; ++it) {
                int task = tid + it * 256;          // 128 rows x 8 float4 groups
                int r = task >> 3, g = task & 7;
                float4 f4 = *(const float4*)(flat + (size_t)toks[r] * Kdim + k0 + g * 4);
                __bf16* dst = &As[r * LDA + g * 4];
                dst[0] = (__bf16)f4.x; dst[1] = (__bf16)f4.y;
                dst[2] = (__bf16)f4.z; dst[3] = (__bf16)f4.w;
            }
        } else {
            // TDM async tile load: one wave issues the descriptor; padding
            // reproduces the LDA=40 row stride the WMMA fragments expect.
            if (tid < 32) {
                const __bf16* gsrc = hdnA + ((size_t)e * cap + m0) * Kdim + k0;
                tdm_load_2d_bf16((unsigned)(uintptr_t)(void*)As, gsrc,
                                 32, 128, (long long)Kdim, 3, 3);
            }
        }
        // ---- stage B tile [32 x 128] transposed into LDS (N-major) ----
        if (k0 + 32 < Kdim) {   // cover latency of next K-step's fp32 weight rows
            int pk = tid >> 5, pg = tid & 31;
            __builtin_prefetch(We + (size_t)(k0 + 32 + pk) * Ncols + n0 + pg * 4, 0, 1);
        }
#pragma unroll
        for (int it = 0; it < 4; ++it) {
            int task = tid + it * 256;              // 32 k-rows x 32 float4 groups
            int k = task >> 5, g = task & 31;
            float4 f4 = *(const float4*)(We + (size_t)(k0 + k) * Ncols + n0 + g * 4);
            Bs[(g * 4 + 0) * LDB + k] = (__bf16)f4.x;
            Bs[(g * 4 + 1) * LDB + k] = (__bf16)f4.y;
            Bs[(g * 4 + 2) * LDB + k] = (__bf16)f4.z;
            Bs[(g * 4 + 3) * LDB + k] = (__bf16)f4.w;
        }
        if (!GEMM1) {
            if (tid < 32) __builtin_amdgcn_s_wait_tensorcnt(0);  // A tile landed
        }
        __syncthreads();

        // ---- fragments per CDNA5 WMMA VGPR layouts ----
        v16bf af[2], bfr[4];
#pragma unroll
        for (int i = 0; i < 2; ++i) {
            const __bf16* rowp = &As[(wm * 32 + i * 16 + lc) * LDA];
            af[i] = ld_frag(rowp + half * 8, rowp + 16 + half * 8);
        }
#pragma unroll
        for (int j = 0; j < 4; ++j) {
            const __bf16* colp = &Bs[(wn * 64 + j * 16 + lc) * LDB];
            bfr[j] = ld_frag(colp + half * 8, colp + 16 + half * 8);
        }
#pragma unroll
        for (int i = 0; i < 2; ++i)
#pragma unroll
            for (int j = 0; j < 4; ++j)
                acc[i][j] = __builtin_amdgcn_wmma_f32_16x16x32_bf16(
                    false, af[i], false, bfr[j], (short)0, acc[i][j], false, false);
        __syncthreads();
    }

    // ---- epilogue ----
#pragma unroll
    for (int i = 0; i < 2; ++i) {
#pragma unroll
        for (int j = 0; j < 4; ++j) {
            int col = n0 + wn * 64 + j * 16 + lc;
            float bv = bias[(size_t)e * Ncols + col];
#pragma unroll
            for (int v = 0; v < 8; ++v) {
                int rowb = wm * 32 + i * 16 + half * 8 + v;   // C layout: VGPR v -> M v / v+8
                float val = acc[i][j][v] + bv;
                if (GEMM1) {
                    val = gelu_exact(val);
                    hdnOut[((size_t)e * cap + m0 + rowb) * Ncols + col] = (__bf16)val;
                } else {
                    atomicAdd(out + (size_t)toks[rowb] * Ncols + col, gts[rowb] * val);
                }
            }
        }
    }
}

// ---------------------------------------------------------------------------
// Host launcher
// ---------------------------------------------------------------------------
extern "C" void kernel_launch(void* const* d_in, const int* in_sizes, int n_in,
                              void* d_out, int out_size, void* d_ws, size_t ws_size,
                              hipStream_t stream)
{
    const float* flat = (const float*)d_in[0];   // [B,T,d] fp32
    const float* Wg   = (const float*)d_in[1];   // [E,d]
    const float* W1   = (const float*)d_in[2];   // [E,d,h]
    const float* b1   = (const float*)d_in[3];   // [E,h]
    const float* W2   = (const float*)d_in[4];   // [E,h,d]
    const float* b2   = (const float*)d_in[5];   // [E,d]

    const int E = NEXP, K = TOPK;
    const int d = in_sizes[1] / E;               // 1024
    const int N = in_sizes[0] / d;               // 16384 tokens
    const int h = in_sizes[2] / (E * d);         // 4096
    const int cap = (2 * N + E - 1) / E;         // 4096

    // workspace carve-out
    char* ws = (char*)d_ws;
    int*   slot_expert = (int*)ws;                 ws += (size_t)N * K * sizeof(int);
    float* slot_score  = (float*)ws;               ws += (size_t)N * K * sizeof(float);
    int*   tokList     = (int*)ws;                 ws += (size_t)E * cap * sizeof(int);
    float* gateList    = (float*)ws;               ws += (size_t)E * cap * sizeof(float);
    ws = (char*)(((uintptr_t)ws + 255) & ~(uintptr_t)255);
    __bf16* hdn        = (__bf16*)ws;              // [E, cap, h] bf16 (~268 MB)

    // zero output (covers final[N,d] and the trailing aux scalar)
    hipMemsetAsync(d_out, 0, (size_t)out_size * sizeof(float), stream);

    router_kernel<<<N / 8, 256, 0, stream>>>(flat, Wg, slot_expert, slot_score, d, N);
    dispatch_kernel<<<E, 1024, 0, stream>>>(slot_expert, slot_score, tokList, gateList,
                                            N * K, cap);
    moe_gemm_kernel<true><<<dim3(h / 128, cap / 128, E), 256, 0, stream>>>(
        flat, nullptr, W1, b1, tokList, gateList, hdn, nullptr, cap, h, d);
    moe_gemm_kernel<false><<<dim3(d / 128, cap / 128, E), 256, 0, stream>>>(
        flat, hdn, W2, b2, tokList, gateList, nullptr, (float*)d_out, cap, d, h);
}
